// DynConv2d_42537356099774
// MI455X (gfx1250) — compile-verified, matrix operand order checked by reading, not run on hardware
//
#include <hip/hip_runtime.h>

// ---------------------------------------------------------------------------
// Dynamic conv2d for MI455X (gfx1250), wave32 + WMMA bf16.
//   x:[32,256,64,64] f32, w1:[65,256], w2:[1024,65], b2:[1024],
//   weight:[4,256,256,3,3]  ->  out:[32,256,64,64] f32
// Phase 1: attention (pool->mlp->softmax/K)      (tiny)
// Phase 2: per-sample aggregated weights in bf16 (38 MB ws, L2 resident)
// Phase 3: conv as 9 shifted GEMMs via v_wmma_f32_16x16x32_bf16 (155 GFLOP)
//          - all 9 weight taps staged in LDS at once -> barrier-free tap loop
//          - 2 output rows per block: 8 WMMA per wave per tap, 1.5 DS/WMMA
//          - GLOBAL_LOAD_ASYNC_TO_LDS_B128 (ASYNCcnt) for weight staging
// ---------------------------------------------------------------------------

typedef __attribute__((ext_vector_type(16))) __bf16 v16bf;
typedef __attribute__((ext_vector_type(8)))  float  v8f;

// async-copy pointer types: builtin wants v4i in addrspace(1)/(3)
typedef int vi4 __attribute__((vector_size(16)));
typedef __attribute__((address_space(1))) vi4* gvi4p;
typedef __attribute__((address_space(3))) vi4* lvi4p;

union FragBF {
  v16bf v;
  uint4 q[2];
};

#if defined(__gfx1250__) && __has_builtin(__builtin_amdgcn_global_load_async_to_lds_b128)
#define USE_ASYNC_COPY 1
#else
#define USE_ASYNC_COPY 0
#endif

__device__ __forceinline__ unsigned short f2bf(float f) {
  unsigned int u = __float_as_uint(f);
  unsigned int r = u + 0x7FFFu + ((u >> 16) & 1u);   // round-to-nearest-even
  return (unsigned short)(r >> 16);
}

// ------------------------------- Phase 1 -----------------------------------
__global__ __launch_bounds__(256) void attn_kernel(
    const float* __restrict__ x, const float* __restrict__ w1,
    const float* __restrict__ w2, const float* __restrict__ b2,
    float* __restrict__ attn)
{
  const int b    = blockIdx.x;
  const int tid  = threadIdx.x;
  const int lane = tid & 31;
  const int wave = tid >> 5;

  __shared__ float pooled[256];
  __shared__ float hbuf[65];

  // global average pool: one wave reduces one channel at a time (coalesced)
  for (int c = wave; c < 256; c += 8) {
    const float* xp = x + (size_t)(b * 256 + c) * 4096;
    float s = 0.f;
    for (int p = lane; p < 4096; p += 32) s += xp[p];
    for (int off = 16; off > 0; off >>= 1) s += __shfl_down(s, off, 32);
    if (lane == 0) pooled[c] = s * (1.0f / 4096.0f);
  }
  __syncthreads();

  if (tid < 65) {
    const float* wp = w1 + tid * 256;
    float s = 0.f;
    for (int c = 0; c < 256; ++c) s += pooled[c] * wp[c];
    hbuf[tid] = s > 0.f ? s : 0.f;           // relu
  }
  __syncthreads();

  {
    const int o = tid;                        // 0..255
    float l[4];
    for (int k = 0; k < 4; ++k) {
      const float* wp = w2 + (k * 256 + o) * 65;
      float s = b2[k * 256 + o];
      for (int j = 0; j < 65; ++j) s += hbuf[j] * wp[j];
      l[k] = s * 2.0f;                        // / TEMP (0.5)
    }
    float m = l[0];
    for (int k = 1; k < 4; ++k) m = fmaxf(m, l[k]);
    float e[4], se = 0.f;
    for (int k = 0; k < 4; ++k) { e[k] = __expf(l[k] - m); se += e[k]; }
    const float inv = 1.0f / se;
    for (int k = 0; k < 4; ++k) attn[(b * 4 + k) * 256 + o] = e[k] * inv;
  }
}

// ------------------------------- Phase 2 -----------------------------------
// agg[b][t][o][c] (bf16), t = 3*i + j tap index, c contiguous.
__global__ __launch_bounds__(256) void aggw_kernel(
    const float* __restrict__ weight, const float* __restrict__ attn,
    unsigned short* __restrict__ agg)
{
  const int gid = blockIdx.x * 256 + threadIdx.x;  // 32*256*256 threads
  const int c = gid & 255;
  const int o = (gid >> 8) & 255;
  const int b = gid >> 16;

  float a[4];
  for (int k = 0; k < 4; ++k) a[k] = attn[(b * 4 + k) * 256 + o];

  const size_t wbase = ((size_t)o * 256 + c) * 9;  // within one expert
  for (int t = 0; t < 9; ++t) {
    float s = 0.f;
    for (int k = 0; k < 4; ++k)
      s += a[k] * weight[(size_t)k * 256 * 256 * 9 + wbase + t];
    agg[((size_t)(b * 9 + t) * 256 + o) * 256 + c] = f2bf(s);
  }
}

// ------------------------------- Phase 3 -----------------------------------
// Block = 128 out-ch (om) x 2 output rows (hb) x sample b.  8 wave32:
// wm = wave>>1 selects 32 out-ch, wn = wave&1 selects which output row.
// Each wave: 2 A-frags x 4 B-frags = 8 WMMA per tap, tap loop barrier-free.
__global__ __launch_bounds__(256) void conv_kernel(
    const float* __restrict__ x, const unsigned short* __restrict__ agg,
    float* __restrict__ out)
{
  const int om   = blockIdx.x;          // 0..1   -> O offset om*128
  const int h0   = blockIdx.y * 2;      // output rows h0, h0+1
  const int b    = blockIdx.z;          // 0..31
  const int tid  = threadIdx.x;
  const int lane = tid & 31;
  const int wave = tid >> 5;
  const int wm   = wave >> 1;           // 0..3 -> M offset wm*32
  const int wn   = wave & 1;            // 0..1 -> output row h0+wn
  const int lm   = lane & 15;
  const int kb   = (lane & 16) ? 8 : 0; // K sub-block per ISA 16-bit layout

  __shared__ __align__(16) unsigned short xs[4 * 66 * 32];    // [xrow][px][c]
  __shared__ __align__(16) unsigned short wt9[9 * 128 * 32];  // [t][o][c]

  v8f acc[2][4];
  for (int i = 0; i < 2; ++i)
    for (int j = 0; j < 4; ++j)
      for (int r = 0; r < 8; ++r) acc[i][j][r] = 0.f;

  for (int c0 = 0; c0 < 256; c0 += 32) {
    __syncthreads();  // all waves done reading previous chunk's LDS

    // ---- stage 4 input rows x 32 channels as bf16, halo columns 0/65 ----
    for (int it = 0; it < 32; ++it) {
      const int flat = it * 256 + tid;        // < 8192
      const int w  = flat & 63;
      const int c  = (flat >> 6) & 31;
      const int di = flat >> 11;              // 0..3 -> input row h0-1+di
      const int hh = h0 + di - 1;
      float v = 0.f;
      if (hh >= 0 && hh < 64)
        v = x[((size_t)(b * 256 + c0 + c) * 64 + hh) * 64 + w];
      xs[(di * 66 + w + 1) * 32 + c] = f2bf(v);
    }
    {                                          // zero halo columns (256 = 4*2*32)
      const int c  = tid & 31;
      const int g  = tid >> 5;                 // 0..7
      const int di = g & 3;
      const int px = (g >= 4) ? 65 : 0;
      xs[(di * 66 + px) * 32 + c] = 0;
    }

    // ---- stage all 9 weight taps: 9*128*32 bf16 = 4608 x 16B vectors ----
    {
      const size_t base = ((size_t)b * 9 * 256 + om * 128) * 256 + c0;
      for (int it = 0; it < 18; ++it) {
        const int vdx = it * 256 + tid;       // 0..4607
        const int t   = vdx >> 9;             // 0..8
        const int r   = vdx & 511;
        const int o   = r >> 2;               // 0..127
        const int col = (r & 3) * 8;
        const size_t gelem = base + ((size_t)t * 256 + o) * 256 + col;
        const int    lelem = (t * 128 + o) * 32 + col;
#if USE_ASYNC_COPY
        __builtin_amdgcn_global_load_async_to_lds_b128(
            (gvi4p)(agg + gelem), (lvi4p)&wt9[lelem], 0, 0);
#else
        *(uint4*)&wt9[lelem] = *(const uint4*)(agg + gelem);
#endif
      }
#if USE_ASYNC_COPY
#if __has_builtin(__builtin_amdgcn_s_wait_asynccnt)
      __builtin_amdgcn_s_wait_asynccnt(0);
#else
      asm volatile("s_wait_asynccnt 0" ::: "memory");
#endif
#endif
    }
    __syncthreads();

    // ---- barrier-free 9-tap GEMM loop ----
    for (int t = 0; t < 9; ++t) {
      const int di = t / 3;
      const int dj = t % 3;                   // px = w_out + dj
      const int wbase = t * 128 * 32;

      FragBF A0, A1;
      const int m0 = wbase + (wm * 32 + lm) * 32;
      A0.q[0] = *(const uint4*)&wt9[m0 + kb];
      A0.q[1] = *(const uint4*)&wt9[m0 + 16 + kb];
      const int m1 = wbase + (wm * 32 + 16 + lm) * 32;
      A1.q[0] = *(const uint4*)&wt9[m1 + kb];
      A1.q[1] = *(const uint4*)&wt9[m1 + 16 + kb];

      const int xrow = wn + di;               // 0..3
      for (int f = 0; f < 4; ++f) {           // 4 x 16-pixel B fragments
        FragBF Bf;
        const int p = (xrow * 66 + f * 16 + lm + dj) * 32;
        Bf.q[0] = *(const uint4*)&xs[p + kb];
        Bf.q[1] = *(const uint4*)&xs[p + 16 + kb];
        acc[0][f] = __builtin_amdgcn_wmma_f32_16x16x32_bf16(
            false, A0.v, false, Bf.v, (short)0, acc[0][f], false, false);
        acc[1][f] = __builtin_amdgcn_wmma_f32_16x16x32_bf16(
            false, A1.v, false, Bf.v, (short)0, acc[1][f], false, false);
      }
    }
  }

  // epilogue: C/D layout -> N = lane&15, M = r + 8*(lane>>4)
  const int mo  = (lane & 16) ? 8 : 0;
  const int row = h0 + wn;
  for (int a = 0; a < 2; ++a)
    for (int f = 0; f < 4; ++f)
      for (int r = 0; r < 8; ++r) {
        const int o = om * 128 + wm * 32 + a * 16 + r + mo;
        const int w = f * 16 + lm;
        out[((size_t)(b * 256 + o) * 64 + row) * 64 + w] = acc[a][f][r];
      }
}

// ---------------------------------------------------------------------------
extern "C" void kernel_launch(void* const* d_in, const int* in_sizes, int n_in,
                              void* d_out, int out_size, void* d_ws, size_t ws_size,
                              hipStream_t stream)
{
  const float* x      = (const float*)d_in[0];
  const float* w1     = (const float*)d_in[1];
  const float* w2     = (const float*)d_in[2];
  const float* b2     = (const float*)d_in[3];
  const float* weight = (const float*)d_in[4];
  float* out = (float*)d_out;

  // workspace layout: attn f32 (128 KB) | agg bf16 (37.75 MB)
  float* attn = (float*)d_ws;
  unsigned short* agg = (unsigned short*)((char*)d_ws + 131072);

  attn_kernel<<<32, 256, 0, stream>>>(x, w1, w2, b2, attn);
  aggw_kernel<<<(32 * 256 * 256) / 256, 256, 0, stream>>>(weight, attn, agg);
  conv_kernel<<<dim3(2, 32, 32), 256, 0, stream>>>(x, agg, out);
}
